// SelfAttention_8229157339627
// MI455X (gfx1250) — matrix operand order repaired and checked by reading, not compiled, not run on hardware
//
#include <hip/hip_runtime.h>
#include <hip/hip_bf16.h>
#include <math.h>
#include <stdint.h>

// ---------------------------------------------------------------------------
// Self-attention, MI455X (gfx1250), wave32.
//  prep : xh = f16(x); WqT/WkT/WvT = f16(W^T)
//  pass1: Qh,Kh = xh@W+b (f16, row-major); Vt = (xh@Wv+b)^T (f16, n-major)
//  pass2: Sc = Qh Kh^T * 1/sqrt(dk)   (fp32 scores)
//  pass3: P = softmax rows -> f16 in place
//  pass4: out = P Vt^T                (fp32)
//
// All-f16 GEMM: A m-major [m][k], B n-major [n][k]; both LDS tiles staged by
// GLOBAL_LOAD_ASYNC_TO_LDS_B128 into double-buffered LDS (ASYNCcnt),
// overlapping DMA with v_wmma. Block tile 128x256, 8 waves as 2x4, each wave
// computes 64x64 = 16 v_wmma_f32_16x16x32_f16 per k-step (16 ds_load_b128).
// ---------------------------------------------------------------------------

typedef __attribute__((ext_vector_type(16))) _Float16 v16h;
typedef __attribute__((ext_vector_type(8)))  _Float16 v8h;
typedef __attribute__((ext_vector_type(8)))  float    v8f;

#define BM 128           // block tile M
#define BN 256           // block tile N
#define BK 32            // k-step (matches 16x16x32)
#define LDT 40           // LDS tile row stride in halves (80B, 16B-aligned)

// --- A fragment: 16x32 f16 (lane m, two contiguous 8-half runs) -----------
__device__ inline v16h load_frag_a(const _Float16* __restrict__ s, int lane) {
  const _Float16* row = s + (lane & 15) * LDT + ((lane >> 4) << 3);
  v8h lo = *(const v8h*)(row);
  v8h hi = *(const v8h*)(row + 16);
  return __builtin_shufflevector(lo, hi, 0, 1, 2, 3, 4, 5, 6, 7,
                                 8, 9, 10, 11, 12, 13, 14, 15);
}

// --- B fragment: 32x16 f16 from n-major tile (lane n, two contiguous 8-half runs) ----
__device__ inline v16h load_frag_b(const _Float16* __restrict__ s, int lane) {
  const _Float16* row = s + (lane & 15) * LDT + ((lane >> 4) << 3);
  v8h lo = *(const v8h*)(row);
  v8h hi = *(const v8h*)(row + 16);
  return __builtin_shufflevector(lo, hi, 0, 1, 2, 3, 4, 5, 6, 7,
                                 8, 9, 10, 11, 12, 13, 14, 15);
}

// --- async row-slab tile copy: ROWS rows x 64B -> LDS (stride LDT halves) --
// ROWS*4 16B chunks over 256 threads; each lane of each wave moves 16B/issue.
template <int ROWS>
__device__ inline void async_copy_tile(uint32_t lds_base,
                                       const _Float16* __restrict__ g,
                                       int ld, int rowBase, int k0, int t) {
  constexpr int PER = (ROWS * 4) / 256;
#pragma unroll
  for (int i = 0; i < PER; ++i) {
    const int c   = t * PER + i;
    const int row = c >> 2;
    const int seg = c & 3;
    const _Float16* src = g + (size_t)(rowBase + row) * ld + k0 + seg * 8;
    const uint32_t  dst = lds_base + (uint32_t)(row * (LDT * 2) + seg * 16);
    asm volatile("global_load_async_to_lds_b128 %0, %1, off"
                 :: "v"(dst), "v"(src) : "memory");
  }
}

__device__ inline void wait_async_all() {
  asm volatile("s_wait_asynccnt 0" ::: "memory");
}

// ---------------------------------------------------------------------------
// C[M,N] = (A x B^T_layout) * scale (+ bias[n]); A m-major, B n-major (f16).
//   CT=false: C[gm*ldc+gn] ; CT=true: C[gn*ldc+gm] (transposed store)
// grid = (N/BN, M/BM, batch), block = 256 (8 wave32s, 2x4 wave grid, 64x64/wave)
// ---------------------------------------------------------------------------
template <typename TC, bool CT, bool HB>
__global__ __launch_bounds__(256)
void gemm_wmma(const _Float16* __restrict__ A, int lda, size_t strideA,
               const _Float16* __restrict__ B, int ldb, size_t strideB,
               TC* __restrict__ C, int ldc, size_t strideC,
               const float* __restrict__ bias, float scale, int K) {
  __shared__ _Float16 As[2][BM * LDT];   // 20 KB
  __shared__ _Float16 Bs[2][BN * LDT];   // 40 KB

  A += strideA * blockIdx.z;
  B += strideB * blockIdx.z;
  C += strideC * blockIdx.z;

  const int t     = threadIdx.x;
  const int lane  = t & 31;
  const int wid   = t >> 5;
  const int waveM = wid >> 2;               // 0..1 -> 64-row slab
  const int waveN = wid & 3;                // 0..3 -> 64-col slab
  const int mBase = blockIdx.y * BM;
  const int nBase = blockIdx.x * BN;

  const uint32_t aBase[2] = {(uint32_t)(uintptr_t)&As[0][0],
                             (uint32_t)(uintptr_t)&As[1][0]};
  const uint32_t bBase[2] = {(uint32_t)(uintptr_t)&Bs[0][0],
                             (uint32_t)(uintptr_t)&Bs[1][0]};

  v8f acc[4][4] = {};

  // prologue: stage first tiles into buffer 0
  async_copy_tile<BM>(aBase[0], A, lda, mBase, 0, t);
  async_copy_tile<BN>(bBase[0], B, ldb, nBase, 0, t);
  wait_async_all();
  __syncthreads();

  int cur = 0;
  for (int k0 = 0; k0 < K; k0 += BK) {
    const int nxt = cur ^ 1;
    if (k0 + BK < K) {                      // kick off next tiles (overlaps WMMA)
      async_copy_tile<BM>(aBase[nxt], A, lda, mBase, k0 + BK, t);
      async_copy_tile<BN>(bBase[nxt], B, ldb, nBase, k0 + BK, t);
    }

    v16h af[4], bf[4];
#pragma unroll
    for (int mi = 0; mi < 4; ++mi)
      af[mi] = load_frag_a(&As[cur][(waveM * 64 + mi * 16) * LDT], lane);
#pragma unroll
    for (int ni = 0; ni < 4; ++ni)
      bf[ni] = load_frag_b(&Bs[cur][(waveN * 64 + ni * 16) * LDT], lane);

#pragma unroll
    for (int mi = 0; mi < 4; ++mi)
#pragma unroll
      for (int ni = 0; ni < 4; ++ni)
        acc[mi][ni] = __builtin_amdgcn_wmma_f32_16x16x32_f16(
            false, af[mi], false, bf[ni], (short)0, acc[mi][ni], false, false);

    wait_async_all();                       // next tiles landed (this wave)
    __syncthreads();                        // all waves done reading cur
    cur = nxt;
  }

  // ---- epilogue: D layout: row = 8*(lane>>4)+r, col = lane&15 ----
  const int mOff = (lane >> 4) * 8;
  const int nOff = lane & 15;
#pragma unroll
  for (int mi = 0; mi < 4; ++mi) {
#pragma unroll
    for (int ni = 0; ni < 4; ++ni) {
      const int gm = mBase + waveM * 64 + mi * 16 + mOff;
      const int gn = nBase + waveN * 64 + ni * 16 + nOff;
      const float bb = HB ? bias[gn] : 0.0f;
#pragma unroll
      for (int r = 0; r < 8; ++r) {
        const float v = acc[mi][ni][r] * scale + bb;
        if (CT)
          C[(size_t)gn * ldc + (gm + r)] = (TC)v;   // contiguous per lane
        else
          C[(size_t)(gm + r) * ldc + gn] = (TC)v;
      }
    }
  }
}

// ---------------------------------------------------------------------------
__global__ __launch_bounds__(256)
void convert_f32_f16(const float* __restrict__ in, _Float16* __restrict__ out) {
  const size_t i = ((size_t)blockIdx.x * 256 + threadIdx.x) * 8;
#pragma unroll
  for (int j = 0; j < 8; ++j)
    out[i + j] = (_Float16)in[i + j];
}

// out[n][k] = (f16) in[k][n] ; 32x32 tiles through LDS; grid (cols/32, rows/32)
__global__ __launch_bounds__(256)
void transpose_convert(const float* __restrict__ in, _Float16* __restrict__ out,
                       int rows, int cols) {
  __shared__ _Float16 tile[32][33];
  const int bx = blockIdx.x * 32;           // col (n) base
  const int by = blockIdx.y * 32;           // row (k) base
  const int tx = threadIdx.x & 31;
  const int ty = threadIdx.x >> 5;          // 0..7
#pragma unroll
  for (int i = 0; i < 32; i += 8)
    tile[ty + i][tx] = (_Float16)in[(size_t)(by + ty + i) * cols + bx + tx];
  __syncthreads();
#pragma unroll
  for (int i = 0; i < 32; i += 8)
    out[(size_t)(bx + ty + i) * rows + by + tx] = tile[tx][ty + i];
}

// ---------------------------------------------------------------------------
__global__ __launch_bounds__(256)
void softmax_rows(const float* __restrict__ S, _Float16* __restrict__ P,
                  int ldS, int ldP) {
  __shared__ float red[256];
  const int t = threadIdx.x;
  const float* srow = S + (size_t)blockIdx.x * ldS;
  _Float16*    prow = P + (size_t)blockIdx.x * ldP;

  float v[8];
  float m = -INFINITY;
#pragma unroll
  for (int j = 0; j < 8; ++j) {
    v[j] = srow[t + 256 * j];
    m = fmaxf(m, v[j]);
  }
  red[t] = m;
  __syncthreads();
  for (int s = 128; s > 0; s >>= 1) {
    if (t < s) red[t] = fmaxf(red[t], red[t + s]);
    __syncthreads();
  }
  m = red[0];
  __syncthreads();

  float sum = 0.0f;
#pragma unroll
  for (int j = 0; j < 8; ++j) {
    v[j] = __expf(v[j] - m);
    sum += v[j];
  }
  red[t] = sum;
  __syncthreads();
  for (int s = 128; s > 0; s >>= 1) {
    if (t < s) red[t] += red[t + s];
    __syncthreads();
  }
  const float inv = 1.0f / red[0];

#pragma unroll
  for (int j = 0; j < 8; ++j)
    prow[t + 256 * j] = (_Float16)(v[j] * inv);
}

// ---------------------------------------------------------------------------
extern "C" void kernel_launch(void* const* d_in, const int* in_sizes, int n_in,
                              void* d_out, int out_size, void* d_ws, size_t ws_size,
                              hipStream_t stream) {
  (void)in_sizes; (void)n_in; (void)out_size; (void)ws_size;

  const float* x  = (const float*)d_in[0];   // [4,2048,1024]
  const float* Wq = (const float*)d_in[1];
  const float* bq = (const float*)d_in[2];
  const float* Wk = (const float*)d_in[3];
  const float* bk = (const float*)d_in[4];
  const float* Wv = (const float*)d_in[5];
  const float* bv = (const float*)d_in[6];
  float* out = (float*)d_out;                // [4,2048,1024] fp32

  const int Bb = 4, S = 2048, D = 1024;
  const int M = Bb * S;                      // 8192

  // workspace layout (136 MB):
  //   [  0, 16M) xh   f16 8192x1024
  //   [ 16, 18M) WqT  f16 1024x1024 (n-major = W^T)
  //   [ 18, 20M) WkT
  //   [ 20, 22M) WvT
  //   [ 24, 40M) Qh   f16 8192x1024 (row-major)
  //   [ 40, 56M) Kh   f16 8192x1024 (row-major; n-major for pass 2)
  //   [ 56, 72M) Vt   f16 1024x8192 (Vt[n][b*2048+s]; n-major for pass 4)
  //   [ 72,136M) Sc   fp32 4x2048x2048 ; softmax rewrites rows as f16 P
  const size_t MB = 1024 * 1024;
  char* ws = (char*)d_ws;
  _Float16* xh  = (_Float16*)(ws);
  _Float16* WqT = (_Float16*)(ws + 16 * MB);
  _Float16* WkT = (_Float16*)(ws + 18 * MB);
  _Float16* WvT = (_Float16*)(ws + 20 * MB);
  _Float16* Qh  = (_Float16*)(ws + 24 * MB);
  _Float16* Kh  = (_Float16*)(ws + 40 * MB);
  _Float16* Vt  = (_Float16*)(ws + 56 * MB);
  float*    Sc  = (float*)   (ws + 72 * MB);
  _Float16* P   = (_Float16*)Sc;

  const dim3 blk(256);
  const float inv_sqrt_dk = 0.03125f;        // 1/sqrt(1024)

  // prep
  convert_f32_f16<<<(M * D) / (256 * 8), blk, 0, stream>>>(x, xh);
  dim3 gt(D / 32, D / 32);
  transpose_convert<<<gt, blk, 0, stream>>>(Wq, WqT, D, D);
  transpose_convert<<<gt, blk, 0, stream>>>(Wk, WkT, D, D);
  transpose_convert<<<gt, blk, 0, stream>>>(Wv, WvT, D, D);

  // pass 1: QKV projections (V stored transposed)
  dim3 g1(D / BN, M / BM, 1);
  gemm_wmma<_Float16, false, true><<<g1, blk, 0, stream>>>(
      xh, D, 0, WqT, D, 0, Qh, D, 0, bq, 1.0f, D);
  gemm_wmma<_Float16, false, true><<<g1, blk, 0, stream>>>(
      xh, D, 0, WkT, D, 0, Kh, D, 0, bk, 1.0f, D);
  gemm_wmma<_Float16, true, true><<<g1, blk, 0, stream>>>(
      xh, D, 0, WvT, D, 0, Vt, M /*ld of Vt row*/, 0, bv, 1.0f, D);

  // pass 2: scores = Q K^T * 1/sqrt(dk)   (Kh row-major == n-major B)
  dim3 g2(S / BN, S / BM, Bb);
  gemm_wmma<float, false, false><<<g2, blk, 0, stream>>>(
      Qh, D, (size_t)S * D, Kh, D, (size_t)S * D,
      Sc, S, (size_t)S * S, nullptr, inv_sqrt_dk, D);

  // pass 3: softmax rows (4*2048), f16 probs in place (ldP = 4096 halves)
  softmax_rows<<<Bb * S, blk, 0, stream>>>(Sc, P, S, 2 * S);

  // pass 4: out = P V  (A = P, lda 4096; B = Vt rows, ldb = M, batch off 2048)
  dim3 g3(D / BN, S / BM, Bb);
  gemm_wmma<float, false, false><<<g3, blk, 0, stream>>>(
      P, 2 * S, (size_t)S * 2 * S, Vt, M, (size_t)S,
      out, D, (size_t)S * D, nullptr, 1.0f, S);
}
